// TranscriptomePredictor_16750372454900
// MI455X (gfx1250) — matrix operand order repaired and checked by reading, not compiled, not run on hardware
//
#include <hip/hip_runtime.h>
#include <hip/hip_bf16.h>

#define BSZ      8
#define GCNT     2048
#define LSEQ     2049
#define ROWS     2056     /* 2048 gene rows + 8 cond rows */
#define D_MODEL  512
#define D_STATE  128
#define D_CONV   4
#define D_INNER  1024
#define NHEADS   16
#define HEADDIM  64
#define CONV_DIM 1280
#define D_IN_PROJ 2320
#define GENE_FEAT 384
#define ID_DIM   128
#define PATH_DIM 128
#define CHR_DIM  64
#define F2       64       /* 2*F_LOCUS */
#define P_DIM    128

typedef __attribute__((ext_vector_type(16))) __bf16 v16bf;
typedef __attribute__((ext_vector_type(8)))  __bf16 v8bf;
typedef __attribute__((ext_vector_type(8)))  float  v8f;

__device__ __forceinline__ unsigned short f2bf(float f) {
  union { float f; unsigned u; } c; c.f = f;
  unsigned u = c.u;
  u += 0x7FFFu + ((u >> 16) & 1u);          // round-to-nearest-even
  return (unsigned short)(u >> 16);
}
__device__ __forceinline__ float siluf(float x) { return x / (1.0f + expf(-x)); }

// Row map into the shared 2056-row activation buffer.
// dir 0 (fwd):  t==0 -> cond row (2048+b), else gene t-1
// dir 1 (bwd):  t==2048 -> cond row, else gene 2047-t   (flipped sequence)
__device__ __forceinline__ int map_row(int dir, int b, int t) {
  if (t < 0) return -1;
  if (dir == 0) return (t == 0) ? (GCNT + b) : (t - 1);
  return (t == LSEQ - 1) ? (GCNT + b) : (GCNT - 1 - t);
}

// ---------------- fp32 -> bf16 conversion ----------------
__global__ void k_cvt(const float* __restrict__ in, unsigned short* __restrict__ out, int n) {
  int i = blockIdx.x * blockDim.x + threadIdx.x;
  if (i < n) out[i] = f2bf(in[i]);
}

// ---------------- gene feature rows (2048 x 384, bf16) ----------------
__global__ void k_build_feat(const float* __restrict__ gid, const float* __restrict__ path,
                             const int* __restrict__ chr_idx, const float* __restrict__ cemb,
                             const float* __restrict__ lf, const float* __restrict__ lw,
                             const float* __restrict__ lb, unsigned short* __restrict__ feat) {
  int g = blockIdx.x, t = threadIdx.x;  // 128 threads
  unsigned short* row = feat + (size_t)g * GENE_FEAT;
  row[t]          = f2bf(gid[(size_t)g * ID_DIM + t]);
  row[ID_DIM + t] = f2bf(path[(size_t)g * PATH_DIM + t]);
  if (t < CHR_DIM) {
    row[ID_DIM + PATH_DIM + t] = f2bf(cemb[(size_t)chr_idx[g] * CHR_DIM + t]);
    float acc = lb[t];
    const float* lfr = lf + (size_t)g * F2;
    for (int j = 0; j < F2; j++) acc += lfr[j] * lw[t * F2 + j];
    float ge = 0.5f * acc * (1.0f + erff(acc * 0.70710678118f));  // exact GELU
    row[ID_DIM + PATH_DIM + CHR_DIM + t] = f2bf(ge);
  }
}

// ---------------- cond feature rows (8 x 384, bf16) ----------------
__global__ void k_build_cond(const int* __restrict__ pidx, const float* __restrict__ pemb,
                             const float* __restrict__ cw, const float* __restrict__ cb,
                             unsigned short* __restrict__ feat) {
  int b = blockIdx.x, t = threadIdx.x;  // 128 threads
  __shared__ float sp[P_DIM];
  if (t < P_DIM) sp[t] = pemb[(size_t)pidx[b] * P_DIM + t];
  __syncthreads();
  unsigned short* row = feat + (size_t)(GCNT + b) * GENE_FEAT;
  for (int c = t; c < GENE_FEAT; c += blockDim.x) {
    float acc = cb[c];
    for (int j = 0; j < P_DIM; j++) acc += sp[j] * cw[(size_t)c * P_DIM + j];
    row[c] = f2bf(acc);
  }
}

// ---------------- WMMA bf16 GEMM:  C[M,N] = A[M,K] @ W[N,K]^T (+bias) ----------------
// Block tile 64x128, 8 waves of 32x32 (2x2 wmma accumulators), K staged 64 at a time.
// As row-major [m][k], Bs n-major [n][k] so every lane fragment is two 16B LDS loads.
#define BM 64
#define BN 128
#define BK 64
#define BKP 72   /* padded row: 144B keeps 16B alignment, kills bank conflicts */

__device__ __forceinline__ v16bf frag16(const unsigned short* p0, const unsigned short* p1) {
  v8bf lo = *(const v8bf*)p0;
  v8bf hi = *(const v8bf*)p1;
  return __builtin_shufflevector(lo, hi, 0, 1, 2, 3, 4, 5, 6, 7,
                                         8, 9, 10, 11, 12, 13, 14, 15);
}

__global__ __launch_bounds__(256)
void k_gemm_bf16(const unsigned short* __restrict__ A, int M, int K, int lda,
                 const unsigned short* __restrict__ W, int N,
                 float* __restrict__ Cf, unsigned short* __restrict__ Cbf,
                 const float* __restrict__ bias, int ldc) {
  __shared__ unsigned short As[BM][BKP];
  __shared__ unsigned short Bs[BN][BKP];
  int mBase = blockIdx.y * BM, nBase = blockIdx.x * BN;
  int tid = threadIdx.x, lane = tid & 31, wave = tid >> 5;
  int mOff = (wave & 1) * 32, nOff = (wave >> 1) * 32;
  int lrow = lane & 15, lhalf = lane >> 4;

  // Per-thread vectorized load coordinates (uint4 = 8 bf16 = 16B)
  int ar = tid >> 3, aq = tid & 7;          // A: 64 rows x 8 quads; 512 uint4, 2 per thread
  int wr = tid >> 3, wq = tid & 7;          // W: 128 rows x 8 quads; 1024 uint4, 4 per thread

  v8f acc[2][2];
  for (int i = 0; i < 2; i++)
    for (int j = 0; j < 2; j++) {
      v8f z = {0.f, 0.f, 0.f, 0.f, 0.f, 0.f, 0.f, 0.f};
      acc[i][j] = z;
    }

  for (int k0 = 0; k0 < K; k0 += BK) {
    // ---- stage A tile (branchless guard: clamp row, zero-select) ----
    #pragma unroll
    for (int it = 0; it < 2; it++) {
      int r = ar + it * 32;
      int gr = mBase + r;
      int grc = (gr < M) ? gr : (M - 1);
      uint4 vv = *(const uint4*)&A[(size_t)grc * lda + k0 + aq * 8];
      if (gr >= M) { vv.x = 0u; vv.y = 0u; vv.z = 0u; vv.w = 0u; }
      *(uint4*)&As[r][aq * 8] = vv;
      if (k0 + BK < K)
        __builtin_prefetch(&A[(size_t)grc * lda + k0 + BK + aq * 8], 0, 0);
    }
    // ---- stage W tile, n-major ----
    #pragma unroll
    for (int it = 0; it < 4; it++) {
      int nn = wr + it * 32;
      int gn = nBase + nn;
      int gnc = (gn < N) ? gn : (N - 1);
      uint4 vv = *(const uint4*)&W[(size_t)gnc * K + k0 + wq * 8];
      if (gn >= N) { vv.x = 0u; vv.y = 0u; vv.z = 0u; vv.w = 0u; }
      *(uint4*)&Bs[nn][wq * 8] = vv;
      if (k0 + BK < K)
        __builtin_prefetch(&W[(size_t)gnc * K + k0 + BK + wq * 8], 0, 0);
    }
    __syncthreads();

    #pragma unroll
    for (int ks = 0; ks < 2; ks++) {             // two 16x16x32 k-subtiles
      int kb = ks * 32;
      v16bf af[2], bfr[2];
      #pragma unroll
      for (int m16 = 0; m16 < 2; m16++) {
        int r = mOff + m16 * 16 + lrow;
        // element i: k = (i<8 ? i : i+8) + lhalf*8  -> two contiguous 8-runs
        af[m16] = frag16(&As[r][kb + lhalf * 8], &As[r][kb + 16 + lhalf * 8]);
      }
      #pragma unroll
      for (int n16 = 0; n16 < 2; n16++) {
        int nn = nOff + n16 * 16 + lrow;
        // element i: k = (i<8 ? i : i+8) + lhalf*8  -> two contiguous 8-runs
        bfr[n16] = frag16(&Bs[nn][kb + lhalf * 8], &Bs[nn][kb + 16 + lhalf * 8]);
      }
      #pragma unroll
      for (int m16 = 0; m16 < 2; m16++)
        #pragma unroll
        for (int n16 = 0; n16 < 2; n16++)
          acc[m16][n16] = __builtin_amdgcn_wmma_f32_16x16x32_bf16(
              false, af[m16], false, bfr[n16], (short)0, acc[m16][n16], false, false);
    }
    __syncthreads();
  }

  for (int m16 = 0; m16 < 2; m16++)
    for (int n16 = 0; n16 < 2; n16++) {
      int col = nBase + nOff + n16 * 16 + lrow;
      if (col >= N) continue;
      #pragma unroll
      for (int j = 0; j < 8; j++) {
        int row = mBase + mOff + m16 * 16 + j + lhalf * 8;
        if (row >= M) continue;
        float v = acc[m16][n16][j];
        if (bias) v += bias[col];
        if (Cbf) Cbf[(size_t)row * ldc + col] = f2bf(v);
        else     Cf [(size_t)row * ldc + col] = v;
      }
    }
}

// ---------------- causal depthwise conv(K=4) + SiLU over xBC ----------------
__global__ void k_conv(const float* __restrict__ zx, const float* __restrict__ cw,
                       const float* __restrict__ cb, float* __restrict__ xc, int dir) {
  int bt = blockIdx.x, b = bt / LSEQ, t = bt % LSEQ;
  int r[4];
  #pragma unroll
  for (int k = 0; k < 4; k++) r[k] = map_row(dir, b, t - 3 + k);
  for (int c = threadIdx.x; c < CONV_DIM; c += blockDim.x) {
    float acc = cb[c];
    #pragma unroll
    for (int k = 0; k < 4; k++)
      if (r[k] >= 0) acc += cw[c * D_CONV + k] * zx[(size_t)r[k] * D_IN_PROJ + D_INNER + c];
    xc[(size_t)bt * CONV_DIM + c] = siluf(acc);
  }
}

// ---------------- dt softplus + dA = exp(-exp(A_log)*dt) ----------------
__global__ void k_dt(const float* __restrict__ zx, const float* __restrict__ dt_bias,
                     const float* __restrict__ A_log, float* __restrict__ dtb,
                     float* __restrict__ dab, int dir) {
  int i = blockIdx.x * blockDim.x + threadIdx.x;
  if (i >= BSZ * LSEQ * NHEADS) return;
  int h = i & 15, bt = i >> 4, b = bt / LSEQ, t = bt % LSEQ;
  int r = map_row(dir, b, t);
  float x = zx[(size_t)r * D_IN_PROJ + D_INNER + CONV_DIM + h] + dt_bias[h];
  float dt = (x > 20.f) ? x : log1pf(expf(x));
  dtb[i] = dt;
  dab[i] = expf(-expf(A_log[h]) * dt);
}

// ---------------- sequential selective-state scan, one block per (b,head) ----------------
// Thread tid: state row p = tid>>2 (0..63), state-col chunk j = tid&3 (32 n's each),
// 64x128 fp32 state lives entirely in VGPRs (32 per thread).
__global__ __launch_bounds__(256)
void k_scan(const float* __restrict__ xc, const float* __restrict__ dtb,
            const float* __restrict__ dab, const float* __restrict__ Dp,
            float* __restrict__ yraw) {
  int bh = blockIdx.x, b = bh >> 4, h = bh & 15;
  int tid = threadIdx.x, p = tid >> 2, j = tid & 3, nb = j * 32;
  __shared__ float sB[D_STATE], sC[D_STATE], sX[HEADDIM];
  float hst[32];
  #pragma unroll
  for (int i = 0; i < 32; i++) hst[i] = 0.f;
  float Dh = Dp[h];
  for (int t = 0; t < LSEQ; t++) {
    size_t rowo = ((size_t)b * LSEQ + t) * CONV_DIM;
    if (tid < D_STATE) sB[tid] = xc[rowo + D_INNER + tid];
    else               sC[tid - D_STATE] = xc[rowo + D_INNER + D_STATE + (tid - D_STATE)];
    if (tid < HEADDIM) sX[tid] = xc[rowo + (size_t)h * HEADDIM + tid];
    __syncthreads();
    int bt16 = ((b * LSEQ) + t) * NHEADS + h;
    float dA = dab[bt16], dt = dtb[bt16];
    float xp = sX[p], s = dt * xp, yp = 0.f;
    #pragma unroll
    for (int i = 0; i < 32; i++) {
      float hv = fmaf(dA, hst[i], s * sB[nb + i]);
      hst[i] = hv;
      yp = fmaf(hv, sC[nb + i], yp);
    }
    yp += __shfl_xor(yp, 1);   // 4 lanes per row are contiguous -> same wave32
    yp += __shfl_xor(yp, 2);
    if (j == 0)
      yraw[((size_t)b * LSEQ + t) * D_INNER + h * HEADDIM + p] = yp + Dh * xp;
    __syncthreads();
  }
}

// ---------------- fold out_w into head_w:  v[j] = sum_c head_w[c]*out_w[c][j] ----------------
__global__ void k_vproj(const float* __restrict__ head_w, const float* __restrict__ out_w,
                        float* __restrict__ v) {
  int jdx = blockIdx.x * blockDim.x + threadIdx.x;
  if (jdx >= D_INNER) return;
  float acc = 0.f;
  for (int c = 0; c < D_MODEL; c++) acc += head_w[c] * out_w[(size_t)c * D_INNER + jdx];
  v[jdx] = acc;
}

// ---------------- gate * RMSNorm * norm_w, then dot with folded head vector ----------------
__global__ void k_gate(const float* __restrict__ yraw, const float* __restrict__ zx,
                       const float* __restrict__ norm_w, const float* __restrict__ v,
                       float* __restrict__ contrib, int dir) {
  int bt = blockIdx.x, b = bt / LSEQ, t = bt % LSEQ;
  int r = map_row(dir, b, t);
  const float* yr = yraw + (size_t)bt * D_INNER;
  const float* zr = zx + (size_t)r * D_IN_PROJ;
  float ss = 0.f, dot = 0.f;
  for (int c = threadIdx.x; c < D_INNER; c += blockDim.x) {
    float g = yr[c] * siluf(zr[c]);
    ss  = fmaf(g, g, ss);
    dot = fmaf(g, norm_w[c] * v[c], dot);
  }
  __shared__ float rs[256], rd[256];
  rs[threadIdx.x] = ss; rd[threadIdx.x] = dot;
  __syncthreads();
  for (int o = 128; o > 0; o >>= 1) {
    if (threadIdx.x < (unsigned)o) { rs[threadIdx.x] += rs[threadIdx.x + o]; rd[threadIdx.x] += rd[threadIdx.x + o]; }
    __syncthreads();
  }
  if (threadIdx.x == 0)
    contrib[bt] = rd[0] * rsqrtf(rs[0] / (float)D_INNER + 1e-5f);
}

// ---------------- combine fwd + flipped bwd + head bias ----------------
__global__ void k_final(const float* __restrict__ cf, const float* __restrict__ cbw,
                        const float* __restrict__ head_b, float* __restrict__ out) {
  int i = blockIdx.x * blockDim.x + threadIdx.x;
  if (i >= BSZ * GCNT) return;
  int b = i / GCNT, g = i % GCNT;
  out[i] = head_b[0] + cf[(size_t)b * LSEQ + (g + 1)] + cbw[(size_t)b * LSEQ + (GCNT - 1 - g)];
}

extern "C" void kernel_launch(void* const* d_in, const int* in_sizes, int n_in,
                              void* d_out, int out_size, void* d_ws, size_t ws_size,
                              hipStream_t stream) {
  (void)in_sizes; (void)n_in; (void)out_size; (void)ws_size;
  const int*   pidx    = (const int*)d_in[0];
  const int*   chr_idx = (const int*)d_in[1];
  const float* locus_f = (const float*)d_in[2];
  const float* pathf   = (const float*)d_in[3];
  const float* pemb    = (const float*)d_in[4];
  const float* gid     = (const float*)d_in[5];
  const float* cemb    = (const float*)d_in[6];
  const float* lw      = (const float*)d_in[7];
  const float* lb      = (const float*)d_in[8];
  const float* condw   = (const float*)d_in[9];
  const float* condb   = (const float*)d_in[10];
  const float* inw     = (const float*)d_in[11];
  const float* inb     = (const float*)d_in[12];
  const float* headw   = (const float*)d_in[13];
  const float* headb   = (const float*)d_in[14];

  char* base = (char*)d_ws;
  size_t cur = 0;
  auto alloc = [&](size_t bytes) -> void* {
    void* p = base + cur;
    cur = (cur + bytes + 255) & ~(size_t)255;
    return p;
  };
  unsigned short* featB = (unsigned short*)alloc((size_t)ROWS * GENE_FEAT * 2);
  unsigned short* inwB  = (unsigned short*)alloc((size_t)D_MODEL * GENE_FEAT * 2);
  unsigned short* seqB  = (unsigned short*)alloc((size_t)ROWS * D_MODEL * 2);
  unsigned short* dinwB = (unsigned short*)alloc((size_t)D_IN_PROJ * D_MODEL * 2);
  float* zx    = (float*)alloc((size_t)ROWS * D_IN_PROJ * 4);
  float* xc    = (float*)alloc((size_t)BSZ * LSEQ * CONV_DIM * 4);
  float* dtb   = (float*)alloc((size_t)BSZ * LSEQ * NHEADS * 4);
  float* dab   = (float*)alloc((size_t)BSZ * LSEQ * NHEADS * 4);
  float* yraw  = (float*)alloc((size_t)BSZ * LSEQ * D_INNER * 4);
  float* vproj = (float*)alloc((size_t)D_INNER * 4);
  float* ctf   = (float*)alloc((size_t)BSZ * LSEQ * 4);
  float* ctb   = (float*)alloc((size_t)BSZ * LSEQ * 4);

  // Stage A: features (2056 unique rows) and shared D_MODEL projection
  k_cvt<<<(D_MODEL * GENE_FEAT + 255) / 256, 256, 0, stream>>>(inw, inwB, D_MODEL * GENE_FEAT);
  k_build_feat<<<GCNT, 128, 0, stream>>>(gid, pathf, chr_idx, cemb, locus_f, lw, lb, featB);
  k_build_cond<<<BSZ, 128, 0, stream>>>(pidx, pemb, condw, condb, featB);
  dim3 g1((D_MODEL + BN - 1) / BN, (ROWS + BM - 1) / BM);
  k_gemm_bf16<<<g1, 256, 0, stream>>>(featB, ROWS, GENE_FEAT, GENE_FEAT,
                                      inwB, D_MODEL, nullptr, seqB, inb, D_MODEL);

  // Stage B: both directions sequentially, reusing the large scratch buffers
  for (int dir = 0; dir < 2; dir++) {
    const float* dinw   = (const float*)d_in[15 + 8 * dir + 0];
    const float* dconvw = (const float*)d_in[15 + 8 * dir + 1];
    const float* dconvb = (const float*)d_in[15 + 8 * dir + 2];
    const float* ddtb   = (const float*)d_in[15 + 8 * dir + 3];
    const float* dAlog  = (const float*)d_in[15 + 8 * dir + 4];
    const float* dD     = (const float*)d_in[15 + 8 * dir + 5];
    const float* dnw    = (const float*)d_in[15 + 8 * dir + 6];
    const float* doutw  = (const float*)d_in[15 + 8 * dir + 7];
    float* ct = dir ? ctb : ctf;

    k_cvt<<<(D_IN_PROJ * D_MODEL + 255) / 256, 256, 0, stream>>>(dinw, dinwB, D_IN_PROJ * D_MODEL);
    dim3 g2((D_IN_PROJ + BN - 1) / BN, (ROWS + BM - 1) / BM);
    k_gemm_bf16<<<g2, 256, 0, stream>>>(seqB, ROWS, D_MODEL, D_MODEL,
                                        dinwB, D_IN_PROJ, zx, nullptr, nullptr, D_IN_PROJ);
    k_conv<<<BSZ * LSEQ, 256, 0, stream>>>(zx, dconvw, dconvb, xc, dir);
    k_dt<<<(BSZ * LSEQ * NHEADS + 255) / 256, 256, 0, stream>>>(zx, ddtb, dAlog, dtb, dab, dir);
    k_scan<<<BSZ * NHEADS, 256, 0, stream>>>(xc, dtb, dab, dD, yraw);
    k_vproj<<<(D_INNER + 255) / 256, 256, 0, stream>>>(headw, doutw, vproj);
    k_gate<<<BSZ * LSEQ, 256, 0, stream>>>(yraw, zx, dnw, vproj, ct, dir);
  }
  k_final<<<(BSZ * GCNT + 255) / 256, 256, 0, stream>>>(ctf, ctb, headb, (float*)d_out);
}
